// MultiHeadAttentionLayer_36979668418616
// MI455X (gfx1250) — matrix operand order repaired and checked by reading, not compiled, not run on hardware
//
#include <hip/hip_runtime.h>
#include <cstdint>

typedef __attribute__((ext_vector_type(16))) _Float16 v16h;
typedef __attribute__((ext_vector_type(8)))  float    v8f;
typedef __attribute__((ext_vector_type(4)))  float    f32x4;
typedef __attribute__((ext_vector_type(2)))  unsigned u32x2;

#define N_NODES   50000
#define N_EDGES   800000

union Frag16 { v16h v; _Float16 h[16]; unsigned u[8]; };
union PackU  { _Float16 h[2]; unsigned u; };

// low 32 bits of a flat shared-aperture address == wave-relative LDS offset
__device__ __forceinline__ unsigned lds_off(const void* p) {
    return (unsigned)(uintptr_t)p;
}
__device__ __forceinline__ void async_copy_b128(unsigned lds, unsigned long long gaddr) {
    asm volatile("global_load_async_to_lds_b128 %0, %1, off"
                 :: "v"(lds), "v"(gaddr) : "memory");
}
__device__ __forceinline__ void wait_async0() {
    asm volatile("s_wait_asynccnt 0" ::: "memory");
}

// ---------------------------------------------------------------------------
// Kernel 0: zero h_out accumulator (in d_out) and z denominators (in ws)
// ---------------------------------------------------------------------------
__global__ void zero_kernel(float* __restrict__ hacc, float* __restrict__ z) {
    size_t i      = (size_t)blockIdx.x * blockDim.x + threadIdx.x;
    size_t stride = (size_t)gridDim.x * blockDim.x;
    for (size_t j = i; j < (size_t)N_NODES * 128; j += stride) hacc[j] = 0.0f;
    for (size_t j = i; j < (size_t)N_NODES * 8;   j += stride) z[j]    = 0.0f;
}

// ---------------------------------------------------------------------------
// Kernel 1: pre-fragment one 128x128 f32 weight into f16 B-fragment order.
// Wp[((ct*4 + ks)*32 + lane)*8 + v] = pack(W[k0][ct*16+n], W[k0+1][ct*16+n])
// with g=lane>>4, n=lane&15, k0 = ks*32 + 8*g + 2*v + (v>=4 ? 8 : 0)   (8192 dwords = 32 KB)
// ---------------------------------------------------------------------------
__global__ __launch_bounds__(256)
void pack_w_kernel(const float* __restrict__ W, unsigned* __restrict__ Wp) {
    int i  = blockIdx.x * 256 + threadIdx.x;   // 0..8191
    int v  = i & 7;
    int L  = (i >> 3) & 31;
    int ks = (i >> 8) & 3;
    int ct = i >> 10;
    int g = L >> 4, n = L & 15;
    int k0 = ks * 32 + 8 * g + 2 * v + ((v >= 4) ? 8 : 0);
    PackU p;
    p.h[0] = (_Float16)W[(size_t)k0 * 128 + ct * 16 + n];
    p.h[1] = (_Float16)W[(size_t)(k0 + 1) * 128 + ct * 16 + n];
    Wp[i] = p.u;
}

// ---------------------------------------------------------------------------
// B-fragment fetch from pre-fragmented weights (2 x b128 per K-step)
// ---------------------------------------------------------------------------
__device__ __forceinline__ void load_bfrag(Frag16& b, const unsigned* __restrict__ Wp,
                                           int head, int ks, int lane) {
    const uint4* pb = (const uint4*)(Wp + (((head * 4 + ks) * 32 + lane) << 3));
    uint4 b0 = pb[0], b1 = pb[1];
    b.u[0] = b0.x; b.u[1] = b0.y; b.u[2] = b0.z; b.u[3] = b0.w;
    b.u[4] = b1.x; b.u[5] = b1.y; b.u[6] = b1.z; b.u[7] = b1.w;
}

// ---------------------------------------------------------------------------
// 16x16 tile GEMM over K=128, A staged as f32 in LDS (cvt on the fly)
// ---------------------------------------------------------------------------
__device__ __forceinline__ v8f tile_gemm128_f32(const float* __restrict__ shA,
                                                const unsigned* __restrict__ Wp,
                                                int head, int lane) {
    const int g = lane >> 4, m = lane & 15;
    v8f c = {};
#pragma unroll
    for (int ks = 0; ks < 4; ++ks) {
        Frag16 a, b;
        const f32x4* pa0 = (const f32x4*)(shA + m * 128 + ks * 32 + 8 * g);
        const f32x4* pa1 = (const f32x4*)(shA + m * 128 + ks * 32 + 16 + 8 * g);
        f32x4 f0 = pa0[0], f1 = pa0[1], f2 = pa1[0], f3 = pa1[1];
#pragma unroll
        for (int j = 0; j < 4; ++j) {
            a.h[j]      = (_Float16)f0[j];
            a.h[4 + j]  = (_Float16)f1[j];
            a.h[8 + j]  = (_Float16)f2[j];
            a.h[12 + j] = (_Float16)f3[j];
        }
        load_bfrag(b, Wp, head, ks, lane);
        c = __builtin_amdgcn_wmma_f32_16x16x32_f16(
                false, a.v, false, b.v, (short)0, c, false, false);
    }
    return c;
}

// ---------------------------------------------------------------------------
// 16x16 tile GEMM over K=128, A pre-converted to f16 in LDS (no cvt here)
// ---------------------------------------------------------------------------
__device__ __forceinline__ v8f tile_gemm128_f16(const _Float16* __restrict__ shA,
                                                const unsigned* __restrict__ Wp,
                                                int head, int lane) {
    const int g = lane >> 4, m = lane & 15;
    const unsigned* pA = (const unsigned*)shA;   // 64 dwords per 128-elem row
    v8f c = {};
#pragma unroll
    for (int ks = 0; ks < 4; ++ks) {
        Frag16 a, b;
#pragma unroll
        for (int v = 0; v < 8; ++v) {   // v<4: k=32ks+8g+2v ; v>=4: +16
            int kb = (v < 4) ? (2 * v + 8 * g) : (16 + 2 * (v - 4) + 8 * g);
            a.u[v] = pA[(m << 6) + ((ks * 32 + kb) >> 1)];
        }
        load_bfrag(b, Wp, head, ks, lane);
        c = __builtin_amdgcn_wmma_f32_16x16x32_f16(
                false, a.v, false, b.v, (short)0, c, false, false);
    }
    return c;
}

// ---------------------------------------------------------------------------
// Kernel 2: node projections. 16 rows/block (3125 exact tiles), 24 waves:
// wave = (matrix 0..2, head 0..7). Q -> Qh[N][128]; K,V interleaved into
// KV[N][128][2] so the edge kernel gathers K and V with one b64.
// h tile staged f32 via async-to-LDS.
// ---------------------------------------------------------------------------
__global__ __launch_bounds__(768)
void node_qkv_kernel(const float* __restrict__ h,
                     const unsigned* __restrict__ WpQ,
                     const unsigned* __restrict__ WpK,
                     const unsigned* __restrict__ WpV,
                     const float* __restrict__ bQ, const float* __restrict__ bK,
                     const float* __restrict__ bV,
                     float* __restrict__ Qh, float* __restrict__ KV)
{
    __shared__ __align__(16) float shH[16 * 128];   // 8 KB
    const int tid  = threadIdx.x;
    const int row0 = blockIdx.x * 16;

    if (tid < 512) {  // 512 lanes x 16 B = 8 KB
        async_copy_b128(lds_off(&shH[tid * 4]),
                        (unsigned long long)(const void*)(h + (size_t)row0 * 128 + tid * 4));
    }
    wait_async0();
    __syncthreads();

    const int wave = __builtin_amdgcn_readfirstlane(tid >> 5);  // SGPR, 0..23
    const int lane = tid & 31;
    const int mat  = wave >> 3;     // 0=Q 1=K 2=V
    const int head = wave & 7;
    const int g    = lane >> 4;
    const int n    = lane & 15;

    const unsigned* Wp = (mat == 0) ? WpQ : (mat == 1) ? WpK : WpV;
    const float*    Bv = (mat == 0) ? bQ  : (mat == 1) ? bK  : bV;

    v8f c = tile_gemm128_f32(shH, Wp, head, lane);

    const float bias = Bv[head * 16 + n];
    if (mat == 0) {
#pragma unroll
        for (int r = 0; r < 8; ++r) {             // C/D: M = r + 8*g, N = n
            int row = row0 + r + 8 * g;
            Qh[(size_t)row * 128 + head * 16 + n] = c[r] + bias;
        }
    } else {
#pragma unroll
        for (int r = 0; r < 8; ++r) {
            int row = row0 + r + 8 * g;
            KV[((size_t)row * 128 + head * 16 + n) * 2 + (mat - 1)] = c[r] + bias;
        }
    }
}

// ---------------------------------------------------------------------------
// Kernel 3: fused edge pipeline. 16 edges/block (50000 exact tiles),
// 8 waves = 8 heads. e tile converted f32->f16 once at staging (NT loads),
// Qe via WMMA, score -> e_out (NT store), exp, scatter-add.
// ---------------------------------------------------------------------------
__global__ __launch_bounds__(256)
void edge_attn_kernel(const float* __restrict__ e,
                      const int*   __restrict__ src, const int* __restrict__ dst,
                      const unsigned* __restrict__ WpQe,
                      const float* __restrict__ bQe,
                      const float* __restrict__ Qh,
                      const float2* __restrict__ KV,   // [N*128] {K,V}
                      float* __restrict__ hacc,        // [N,128] accum (in d_out)
                      float* __restrict__ z,           // [N,8]
                      float* __restrict__ e_out)       // [E,128]
{
    __shared__ __align__(8) _Float16 shE[16 * 128];    // 4 KB (f16)
    const int tid = threadIdx.x;
    const int e0  = blockIdx.x * 16;

    {   // stage + convert: 2 x b128 NT loads, 4 cvt_pk, 2 x b64 LDS stores
        const f32x4* gsrc = (const f32x4*)(e + (size_t)e0 * 128);
        u32x2* sh64 = (u32x2*)shE;
#pragma unroll
        for (int t = 0; t < 2; ++t) {
            int i = tid + t * 256;                 // float4 index 0..511
            f32x4 f = __builtin_nontemporal_load(&gsrc[i]);
            PackU p0, p1;
            p0.h[0] = (_Float16)f[0]; p0.h[1] = (_Float16)f[1];
            p1.h[0] = (_Float16)f[2]; p1.h[1] = (_Float16)f[3];
            u32x2 w; w[0] = p0.u; w[1] = p1.u;
            sh64[i] = w;
        }
    }
    __syncthreads();

    const int head = __builtin_amdgcn_readfirstlane(tid >> 5);  // SGPR, 0..7
    const int lane = tid & 31;
    const int g    = lane >> 4;
    const int d    = lane & 15;                    // dim within head

    v8f c = tile_gemm128_f16(shE, WpQe, head, lane);
    const float bias = bQe[head * 16 + d];

    // hoist the 8 src/dst indices for this half-wave (consecutive edges)
    const int base = e0 + 8 * g;
    int4 s0 = *(const int4*)(src + base);
    int4 s1 = *(const int4*)(src + base + 4);
    int4 d0 = *(const int4*)(dst + base);
    int4 d1 = *(const int4*)(dst + base + 4);
    int sv[8] = {s0.x, s0.y, s0.z, s0.w, s1.x, s1.y, s1.z, s1.w};
    int dv[8] = {d0.x, d0.y, d0.z, d0.w, d1.x, d1.y, d1.z, d1.w};

#pragma unroll
    for (int r = 0; r < 8; ++r) {
        const int   edge = base + r;               // C/D row M = r + 8*g
        const float qe   = c[r] + bias;
        const int   sa   = sv[r];
        const int   da   = dv[r];
        const float2 kv2 = KV[(size_t)sa * 128 + head * 16 + d];  // {K,V}
        const float  qv  = Qh[(size_t)da * 128 + head * 16 + d];

        float sc = kv2.x * qv * 0.25f * qe;        // / sqrt(D=16)
        __builtin_nontemporal_store(sc, &e_out[(size_t)edge * 128 + head * 16 + d]);

        // per-head sum over d: reduce within each 16-lane half of the wave32
        float red = sc;
        red += __shfl_xor(red, 1, 16);
        red += __shfl_xor(red, 2, 16);
        red += __shfl_xor(red, 4, 16);
        red += __shfl_xor(red, 8, 16);

        float s = __expf(fminf(fmaxf(red, -5.0f), 5.0f));
        atomicAdd(&hacc[(size_t)da * 128 + head * 16 + d], kv2.y * s);
        if (d == 0) atomicAdd(&z[(size_t)da * 8 + head], s);
    }
}

// ---------------------------------------------------------------------------
// Kernel 4: h_out = accum / (z + 1e-6), in place in d_out
// ---------------------------------------------------------------------------
__global__ void finalize_kernel(float* __restrict__ hacc,
                                const float* __restrict__ z) {
    size_t i = (size_t)blockIdx.x * blockDim.x + threadIdx.x;
    if (i < (size_t)N_NODES * 128) {
        size_t nidx = i >> 7;
        int    head = (int)((i >> 4) & 7);
        hacc[i] = hacc[i] / (z[nidx * 8 + head] + 1e-6f);
    }
}

// ---------------------------------------------------------------------------
extern "C" void kernel_launch(void* const* d_in, const int* in_sizes, int n_in,
                              void* d_out, int out_size, void* d_ws, size_t ws_size,
                              hipStream_t stream) {
    (void)in_sizes; (void)n_in; (void)out_size; (void)ws_size;
    const float* h   = (const float*)d_in[0];
    const float* e   = (const float*)d_in[1];
    const int*   src = (const int*)d_in[2];
    const int*   dst = (const int*)d_in[3];
    const float* WQ  = (const float*)d_in[4];
    const float* bQ  = (const float*)d_in[5];
    const float* WK  = (const float*)d_in[6];
    const float* bK  = (const float*)d_in[7];
    const float* WV  = (const float*)d_in[8];
    const float* bV  = (const float*)d_in[9];
    const float* WQe = (const float*)d_in[10];
    const float* bQe = (const float*)d_in[11];

    float* out   = (float*)d_out;
    float* h_out = out;                              // [N,128]
    float* e_out = out + (size_t)N_NODES * 128;      // [E,128]

    float* ws = (float*)d_ws;
    float* Qh = ws;                                  // [N,128]
    float* KV = Qh + (size_t)N_NODES * 128;          // [N,128,2] interleaved K,V
    float* zd = KV + (size_t)N_NODES * 256;          // [N,8]
    unsigned* WpQ  = (unsigned*)(zd + (size_t)N_NODES * 8);
    unsigned* WpK  = WpQ + 8192;
    unsigned* WpV  = WpK + 8192;
    unsigned* WpQe = WpV + 8192;

    zero_kernel<<<2048, 256, 0, stream>>>(h_out, zd);
    pack_w_kernel<<<32, 256, 0, stream>>>(WQ,  WpQ);
    pack_w_kernel<<<32, 256, 0, stream>>>(WK,  WpK);
    pack_w_kernel<<<32, 256, 0, stream>>>(WV,  WpV);
    pack_w_kernel<<<32, 256, 0, stream>>>(WQe, WpQe);
    node_qkv_kernel<<<N_NODES / 16, 768, 0, stream>>>(
        h, WpQ, WpK, WpV, bQ, bK, bV, Qh, KV);
    edge_attn_kernel<<<N_EDGES / 16, 256, 0, stream>>>(
        e, src, dst, WpQe, bQe, Qh, (const float2*)KV, h_out, zd, e_out);
    finalize_kernel<<<((size_t)N_NODES * 128 + 255) / 256, 256, 0, stream>>>(
        h_out, zd);
}
